// HybridCortexBlock_19670950216038
// MI455X (gfx1250) — matrix operand order repaired and checked by reading, not compiled, not run on hardware
//
#include <hip/hip_runtime.h>
#include <stdint.h>

// ---------------------------------------------------------------- types
typedef __attribute__((ext_vector_type(16))) __bf16 v16bf;
typedef __attribute__((ext_vector_type(8)))  float  v8f;
typedef __attribute__((ext_vector_type(4)))  unsigned int u32x4;
typedef __attribute__((ext_vector_type(8)))  int i32x8;
typedef __attribute__((ext_vector_type(4)))  int i32x4;

union FragA { v16bf v; uint16_t h[16]; uint32_t w[8]; };
union FragC { v8f   v; float f[8]; };

__device__ __forceinline__ uint16_t f2bf(float f) {
    union { float f; uint32_t u; } x; x.f = f;
    uint32_t u = x.u;
    uint32_t r = u + 0x7FFFu + ((u >> 16) & 1u);
    if ((u & 0x7F800000u) == 0x7F800000u) r = u;   // inf/nan passthrough
    return (uint16_t)(r >> 16);
}

#if __has_builtin(__builtin_amdgcn_tensor_load_to_lds)
#define HAVE_TDM 1
#else
#define HAVE_TDM 0
#endif

#if HAVE_TDM
// Issue one TDM 2D tile load: tile_w contiguous bf16 elems per row, tile_h rows,
// row stride = stride_elems. Descriptor per CDNA5 ISA.
// This toolchain exposes the 6-arg builtin: (u32x4, i32x8, i32x4, i32x4, i32x8, cpol).
__device__ __forceinline__ void tdm_load_2d(uint32_t lds_off, const uint16_t* gptr,
                                            uint32_t tile_w, uint32_t tile_h,
                                            uint64_t stride_elems,
                                            uint32_t tensor_d0, uint32_t tensor_d1) {
    union { u32x4 v; uint32_t w[4]; } g0;
    union { i32x8 v; uint32_t w[8]; } g1, g4;
    union { i32x4 v; uint32_t w[4]; } g2;
    uint64_t ga = (uint64_t)(uintptr_t)gptr;
    g0.w[0] = 1u;                                   // count=1, not restore/store, no gather
    g0.w[1] = lds_off;                              // lds_addr (bytes)
    g0.w[2] = (uint32_t)ga;                         // global_addr[31:0]
    g0.w[3] = (uint32_t)((ga >> 32) & 0x01FFFFFFu)  // global_addr[56:32]
            | (2u << 30);                           // type=2 ("image")
    g1.w[0] = (1u << 16);                           // data_size=1 -> 2 bytes
    g1.w[1] = (tensor_d0 & 0xFFFFu) << 16;          // tensor_dim0[15:0]
    g1.w[2] = ((tensor_d0 >> 16) & 0xFFFFu)         // tensor_dim0[31:16]
            | ((tensor_d1 & 0xFFFFu) << 16);        // tensor_dim1[15:0]
    g1.w[3] = ((tensor_d1 >> 16) & 0xFFFFu)         // tensor_dim1[31:16]
            | (tile_w << 16);                       // tile_dim0
    g1.w[4] = tile_h;                               // tile_dim1 (tile_dim2=0)
    g1.w[5] = (uint32_t)stride_elems;               // tensor_dim0_stride[31:0]
    g1.w[6] = (uint32_t)(stride_elems >> 32) & 0xFFFFu; // stride[47:32]
    g1.w[7] = 0;
    g2.w[0] = 0; g2.w[1] = 0; g2.w[2] = 0; g2.w[3] = 0;
#pragma unroll
    for (int i = 0; i < 8; ++i) g4.w[i] = 0;
    __builtin_amdgcn_tensor_load_to_lds(g0.v, g1.v, g2.v, g2.v, g4.v, 0);
}
#endif

// ---------------------------------------------------------------- converts
__global__ __launch_bounds__(256) void k_convert(const float* __restrict__ in,
                                                 uint16_t* __restrict__ out, size_t n) {
    size_t i = (size_t)blockIdx.x * 256 + threadIdx.x;
    if (i < n) out[i] = f2bf(in[i]);
}

// out[n*K + k] = bf16(in[k*N + n])  (weight transpose, tiled through LDS)
__global__ __launch_bounds__(256) void k_convert_t(const float* __restrict__ in,
                                                   uint16_t* __restrict__ out,
                                                   int K, int N) {
    __shared__ float tile[32][33];
    int kb = blockIdx.x * 32, nb = blockIdx.y * 32;
    int tx = threadIdx.x & 31, ty = threadIdx.x >> 5;   // 32 x 8
    for (int i = ty; i < 32; i += 8)
        tile[i][tx] = in[(size_t)(kb + i) * N + nb + tx];
    __syncthreads();
    for (int i = ty; i < 32; i += 8)
        out[(size_t)(nb + i) * K + kb + tx] = f2bf(tile[tx][i]);
}

// ---------------------------------------------------------------- generic WMMA GEMM
// C[M,N] = A[M,K](bf16,row-major) * Bt[N,K](bf16, pre-transposed weights) + bias[N].
// act: 0=none, 1=*0.125, 2=softplus, 3=clip(-10,10)->log_sigmoid
// Tiles move global->LDS via the Tensor Data Mover (wave 0 issues the DMA).
__global__ __launch_bounds__(256) void k_gemm(
    const uint16_t* __restrict__ A, const uint16_t* __restrict__ Bt,
    const float* __restrict__ bias, float* __restrict__ outF,
    uint16_t* __restrict__ outBF,
    int M, int N, int K, int lda, int ldb, int ldcF, int ldcBF, int act)
{
    __shared__ uint16_t As[64][32];   // A tile [m][k]
    __shared__ uint16_t Bs[64][32];   // B tile [n][k]

    const int tid  = threadIdx.x;
    const int wave = tid >> 5, lane = tid & 31;
    const int lh   = lane >> 4, ln = lane & 15;
    const int tm   = wave & 3;
    const int tnb  = (wave >> 2) * 2;
    const int m0   = blockIdx.y * 64;
    const int n0   = blockIdx.x * 64;

    FragC acc0, acc1;
#pragma unroll
    for (int i = 0; i < 8; ++i) { acc0.f[i] = 0.f; acc1.f[i] = 0.f; }

    for (int k0 = 0; k0 < K; k0 += 32) {
#if HAVE_TDM
        if (wave == 0) {
            tdm_load_2d((uint32_t)(uintptr_t)&As[0][0],
                        A + (size_t)m0 * lda + k0, 32u, 64u, (uint64_t)lda,
                        (uint32_t)K, (uint32_t)M);
            tdm_load_2d((uint32_t)(uintptr_t)&Bs[0][0],
                        Bt + (size_t)n0 * ldb + k0, 32u, 64u, (uint64_t)ldb,
                        (uint32_t)K, (uint32_t)N);
            __builtin_amdgcn_s_wait_tensorcnt(0);
        }
#else
        {
            int row = tid >> 2, kc = (tid & 3) * 8;
            *(uint4*)&As[row][kc] = *(const uint4*)(A + (size_t)(m0 + row) * lda + k0 + kc);
            *(uint4*)&Bs[row][kc] = *(const uint4*)(Bt + (size_t)(n0 + row) * ldb + k0 + kc);
        }
#endif
        __syncthreads();

        FragA a;
        {   // A fragment: row m = tm*16+ln; elements e -> K = 16*lh + e (contiguous)
            int row = tm * 16 + ln;
#pragma unroll
            for (int j = 0; j < 8; ++j) {
                int kk = 16 * lh + 2 * j;
                a.w[j] = *(const uint32_t*)&As[row][kk];
            }
        }
        FragA b0, b1;
        {   // B fragment: col n; elements e -> K = 16*lh + e (contiguous)
            const uint32_t* p0 = (const uint32_t*)&Bs[tnb * 16 + ln][16 * lh];
            const uint32_t* p1 = (const uint32_t*)&Bs[tnb * 16 + 16 + ln][16 * lh];
#pragma unroll
            for (int j = 0; j < 8; ++j) { b0.w[j] = p0[j]; b1.w[j] = p1[j]; }
        }
        acc0.v = __builtin_amdgcn_wmma_f32_16x16x32_bf16(false, a.v, false, b0.v,
                                                         (short)0, acc0.v, false, false);
        acc1.v = __builtin_amdgcn_wmma_f32_16x16x32_bf16(false, a.v, false, b1.v,
                                                         (short)0, acc1.v, false, false);
        __syncthreads();
    }

#pragma unroll
    for (int s = 0; s < 2; ++s) {
        FragC& acc = s ? acc1 : acc0;
        int n = n0 + (tnb + s) * 16 + ln;
        float bv = bias ? bias[n] : 0.f;
#pragma unroll
        for (int v = 0; v < 8; ++v) {
            int m = m0 + tm * 16 + 8 * lh + v;
            float val = acc.f[v] + bv;
            if (act == 1) val *= 0.125f;
            else if (act == 2) val = (val > 20.f) ? val : log1pf(expf(val));
            else if (act == 3) {
                float c = fminf(fmaxf(val, -10.f), 10.f);
                val = -log1pf(expf(-c));
            }
            if (outF)  outF [(size_t)m * ldcF  + n] = val;
            if (outBF) outBF[(size_t)m * ldcBF + n] = f2bf(val);
        }
    }
}

// ---------------------------------------------------------------- sliding-window attention
// One wave per (b, h, 16-query tile). Flash-style online softmax; S and PV via WMMA.
#define ATT_WIN 128
__global__ __launch_bounds__(128) void k_attn(
    const uint16_t* __restrict__ qkv,   // (B,T,3D) bf16
    uint16_t* __restrict__ outp,        // (B,T,D)  bf16
    int T, int H, int D3, int D)
{
    __shared__ uint16_t Pld[4][16][32]; // per-wave P tile (A layout staging)
    __shared__ uint16_t Vt [4][64][32]; // per-wave V tile transposed [dd][k]

    const int tid = threadIdx.x;
    const int wv = tid >> 5, lane = tid & 31, lh = lane >> 4, ln = lane & 15;
    const int gw = blockIdx.x * 4 + wv;
    const int qtPerBH = T / 16;
    const int bh = gw / qtPerBH, qt = gw % qtPerBH;
    const int b = bh / H, h = bh % H;
    const int q0 = qt * 16;

    const uint16_t* qbase = qkv + (size_t)b * T * D3 + h * 64;
    const uint16_t* kbase = qbase + D;
    const uint16_t* vbase = qbase + 2 * D;

    FragA qf[2];
    {
        const uint16_t* qr = qbase + (size_t)(q0 + ln) * D3;
#pragma unroll
        for (int ks = 0; ks < 2; ++ks)
#pragma unroll
            for (int j = 0; j < 8; ++j) {
                int kk = 32 * ks + 16 * lh + 2 * j;
                qf[ks].w[j] = *(const uint32_t*)(qr + kk);
            }
    }

    FragC o[4];
#pragma unroll
    for (int nt = 0; nt < 4; ++nt)
#pragma unroll
        for (int i = 0; i < 8; ++i) o[nt].f[i] = 0.f;
    float mrow[8], ssum[8];
#pragma unroll
    for (int v = 0; v < 8; ++v) { mrow[v] = -1e30f; ssum[v] = 0.f; }

    int jstart = q0 - ATT_WIN; if (jstart < 0) jstart = 0;
    jstart &= ~31;

    for (int j0 = jstart; j0 <= q0 + 15; j0 += 32) {
        FragC s0, s1;
#pragma unroll
        for (int i = 0; i < 8; ++i) { s0.f[i] = 0.f; s1.f[i] = 0.f; }

        int kr0 = j0 + ln;      if (kr0 > T - 1) kr0 = T - 1;
        int kr1 = j0 + 16 + ln; if (kr1 > T - 1) kr1 = T - 1;
#pragma unroll
        for (int ks = 0; ks < 2; ++ks) {
            FragA kf0, kf1;
            const uint32_t* p0 = (const uint32_t*)(kbase + (size_t)kr0 * D3 + 32 * ks + 16 * lh);
            const uint32_t* p1 = (const uint32_t*)(kbase + (size_t)kr1 * D3 + 32 * ks + 16 * lh);
#pragma unroll
            for (int j = 0; j < 8; ++j) { kf0.w[j] = p0[j]; kf1.w[j] = p1[j]; }
            s0.v = __builtin_amdgcn_wmma_f32_16x16x32_bf16(false, qf[ks].v, false, kf0.v,
                                                           (short)0, s0.v, false, false);
            s1.v = __builtin_amdgcn_wmma_f32_16x16x32_bf16(false, qf[ks].v, false, kf1.v,
                                                           (short)0, s1.v, false, false);
        }

        float rescale[8];
#pragma unroll
        for (int v = 0; v < 8; ++v) {
            int qi = q0 + 8 * lh + v;
            int ja = j0 + ln, jb = j0 + 16 + ln;
            float a0 = s0.f[v] * 0.125f, a1 = s1.f[v] * 0.125f;
            if (ja > qi || qi - ja > ATT_WIN) a0 = -1e30f;
            if (jb > qi || qi - jb > ATT_WIN) a1 = -1e30f;
            float mx = fmaxf(a0, a1);
            mx = fmaxf(mx, __shfl_xor(mx, 1));
            mx = fmaxf(mx, __shfl_xor(mx, 2));
            mx = fmaxf(mx, __shfl_xor(mx, 4));
            mx = fmaxf(mx, __shfl_xor(mx, 8));
            float mnew = fmaxf(mrow[v], mx);
            rescale[v] = expf(mrow[v] - mnew);
            mrow[v] = mnew;
            float p0 = (a0 <= -1e29f) ? 0.f : expf(a0 - mnew);
            float p1 = (a1 <= -1e29f) ? 0.f : expf(a1 - mnew);
            s0.f[v] = p0; s1.f[v] = p1;
            float rs = p0 + p1;
            rs += __shfl_xor(rs, 1); rs += __shfl_xor(rs, 2);
            rs += __shfl_xor(rs, 4); rs += __shfl_xor(rs, 8);
            ssum[v] = ssum[v] * rescale[v] + rs;
#pragma unroll
            for (int nt = 0; nt < 4; ++nt) o[nt].f[v] *= rescale[v];
        }

        // stage P (C layout -> LDS) and V tile (transposed) for re-fragmentation
#pragma unroll
        for (int v = 0; v < 8; ++v) {
            int row = 8 * lh + v;
            Pld[wv][row][ln]      = f2bf(s0.f[v]);
            Pld[wv][row][ln + 16] = f2bf(s1.f[v]);
        }
        {
            int krow = j0 + lane; if (krow > T - 1) krow = T - 1;
            const uint16_t* vr = vbase + (size_t)krow * D3;
#pragma unroll
            for (int c = 0; c < 8; ++c) {
                union { uint4 q; uint16_t s[8]; } d;
                d.q = *(const uint4*)(vr + c * 8);
#pragma unroll
                for (int i = 0; i < 8; ++i) Vt[wv][c * 8 + i][lane] = d.s[i];
            }
        }
        asm volatile("s_wait_dscnt 0" ::: "memory");   // wave-local LDS fence
        __builtin_amdgcn_wave_barrier();

        FragA pf;
#pragma unroll
        for (int j = 0; j < 8; ++j) {
            int kk = 16 * lh + 2 * j;
            pf.w[j] = *(const uint32_t*)&Pld[wv][ln][kk];
        }
#pragma unroll
        for (int nt = 0; nt < 4; ++nt) {
            FragA vf;
            const uint32_t* p = (const uint32_t*)&Vt[wv][nt * 16 + ln][16 * lh];
#pragma unroll
            for (int j = 0; j < 8; ++j) vf.w[j] = p[j];
            o[nt].v = __builtin_amdgcn_wmma_f32_16x16x32_bf16(false, pf.v, false, vf.v,
                                                              (short)0, o[nt].v, false, false);
        }
        __builtin_amdgcn_wave_barrier();
    }

#pragma unroll
    for (int nt = 0; nt < 4; ++nt)
#pragma unroll
        for (int v = 0; v < 8; ++v) {
            int row = q0 + 8 * lh + v;
            int dd  = nt * 16 + ln;
            outp[((size_t)(b * T + row)) * D + h * 64 + dd] = f2bf(o[nt].f[v] / ssum[v]);
        }
}

// ---------------------------------------------------------------- delta-rule chunked scan
__global__ __launch_bounds__(256) void k_delta(
    const float* __restrict__ qp, const float* __restrict__ kp,
    const float* __restrict__ vp, const float* __restrict__ betap,
    const float* __restrict__ logap, const float* __restrict__ state0,
    float* __restrict__ memsmall, float* __restrict__ state_out,
    int Tp, int H, int D)
{
    __shared__ float S[64][64];
    __shared__ float qc[32][64], kc[32][64], vc[32][64], bcs[32][64], lc[32][64];
    __shared__ float attn[32][32];
    __shared__ float rowsum[32], bmean[32], mch[32], totpart[64];
    __shared__ float totsh;

    const int tid = threadIdx.x;
    const int b = blockIdx.x / H, h = blockIdx.x % H;

    const float* st = state0 + (size_t)(b * H + h) * 64 * 64;
    for (int i = tid; i < 4096; i += 256) S[i >> 6][i & 63] = st[i];
    __syncthreads();

    const int nchunks = Tp / 32;
    for (int t = 0; t < nchunks; ++t) {
        for (int i = tid; i < 2048; i += 256) {
            int c = i >> 6, dd = i & 63;
            size_t idx = (size_t)(b * Tp + t * 32 + c) * D + h * 64 + dd;
            qc[c][dd] = qp[idx]; kc[c][dd] = kp[idx]; vc[c][dd] = vp[idx];
            bcs[c][dd] = betap[idx]; lc[c][dd] = logap[idx];
        }
        __syncthreads();

        if (tid < 32) {
            float rs = 0.f, bs = 0.f;
            for (int dd = 0; dd < 64; ++dd) { rs += lc[tid][dd]; bs += bcs[tid][dd]; }
            rowsum[tid] = rs; bmean[tid] = bs * (1.f / 64.f);
        }
        if (tid >= 64 && tid < 128) {
            int dd = tid - 64;
            float cs = 0.f;
            for (int c = 0; c < 32; ++c) cs += lc[c][dd];
            totpart[dd] = expf(cs);
        }
        __syncthreads();
        if (tid < 32) {
            float acc = 0.f;
            for (int c = 0; c <= tid; ++c) acc += rowsum[c];
            mch[tid] = acc * (1.f / 64.f);
        }
        if (tid == 64) {
            float s = 0.f;
            for (int dd = 0; dd < 64; ++dd) s += totpart[dd];
            totsh = s * (1.f / 64.f);
        }
        __syncthreads();

        for (int i = tid; i < 1024; i += 256) {
            int ii = i >> 5, jj = i & 31;
            float a = 0.f;
            if (jj <= ii) {
                float dot = 0.f;
                for (int dd = 0; dd < 64; ++dd) dot += qc[ii][dd] * kc[jj][dd];
                a = dot * expf(mch[ii] - mch[jj]);
            }
            attn[ii][jj] = a;
        }
        __syncthreads();

        for (int i = tid; i < 2048; i += 256) {
            int c = i >> 6, dd = i & 63;
            float y = 0.f;
            for (int j = 0; j <= c; ++j) y += attn[c][j] * vc[j][dd] * bcs[j][dd];
            float mv = 0.f;
            for (int kk = 0; kk < 64; ++kk) mv += qc[c][kk] * S[kk][dd];
            mv *= expf(mch[c]);
            memsmall[(size_t)(b * Tp + t * 32 + c) * D + h * 64 + dd] = y + mv;
        }
        __syncthreads();

        float tot = totsh;
        for (int i = tid; i < 4096; i += 256) {
            int kk = i >> 6, vv = i & 63;
            float kv = 0.f;
            for (int c = 0; c < 32; ++c) kv += kc[c][kk] * vc[c][vv] * bmean[c];
            S[kk][vv] = S[kk][vv] * tot + kv;
        }
        __syncthreads();
    }

    float* so = state_out + (size_t)(b * H + h) * 64 * 64;
    for (int i = tid; i < 4096; i += 256) so[i] = S[i >> 6][i & 63];
}

// ---------------------------------------------------------------- mem expand + combine
__global__ __launch_bounds__(256) void k_expand_mem(
    const float* __restrict__ ms, float* __restrict__ mem, uint16_t* __restrict__ concatb,
    int Bsz, int T, int D, int r)
{
    size_t i = (size_t)blockIdx.x * 256 + threadIdx.x;
    size_t n = (size_t)Bsz * T * D;
    if (i >= n) return;
    int dd = (int)(i % D);
    size_t bt = i / D;
    int t = (int)(bt % T), b = (int)(bt / T);
    float v = 0.f;
    if (t >= r) v = ms[(size_t)(b * (T / r) + (t - r) / r) * D + dd];
    mem[i] = v;
    concatb[(size_t)(b * T + t) * (2 * D) + D + dd] = f2bf(v);
}

__global__ __launch_bounds__(256) void k_combine(
    const float* __restrict__ gl, const float* __restrict__ attnf,
    const float* __restrict__ mem, uint16_t* __restrict__ comb, size_t n)
{
    size_t i = (size_t)blockIdx.x * 256 + threadIdx.x;
    if (i >= n) return;
    float g = 1.f / (1.f + expf(-gl[i]));
    comb[i] = f2bf(g * attnf[i] + (1.f - g) * mem[i]);
}

// ---------------------------------------------------------------- host
extern "C" void kernel_launch(void* const* d_in, const int* in_sizes, int n_in,
                              void* d_out, int out_size, void* d_ws, size_t ws_size,
                              hipStream_t stream) {
    (void)in_sizes; (void)n_in; (void)out_size;
    const float* x      = (const float*)d_in[0];
    const float* state  = (const float*)d_in[1];
    const float* Wqkv   = (const float*)d_in[2];  const float* bqkv  = (const float*)d_in[3];
    const float* Wo     = (const float*)d_in[4];  const float* bo    = (const float*)d_in[5];
    const float* Wc     = (const float*)d_in[6];  const float* bcv   = (const float*)d_in[7];
    const float* Wq     = (const float*)d_in[8];  const float* bq    = (const float*)d_in[9];
    const float* Wk     = (const float*)d_in[10]; const float* bk    = (const float*)d_in[11];
    const float* Wv     = (const float*)d_in[12]; const float* bv    = (const float*)d_in[13];
    const float* Wbeta  = (const float*)d_in[14]; const float* bbeta = (const float*)d_in[15];
    const float* Walpha = (const float*)d_in[16]; const float* balpha= (const float*)d_in[17];
    const float* Wg     = (const float*)d_in[18]; const float* bg    = (const float*)d_in[19];
    const float* Wout   = (const float*)d_in[20]; const float* bout  = (const float*)d_in[21];

    constexpr int Bsz = 2, T = 2048, D = 1024, H = 16, R = 4;
    constexpr int Tp = T / R;                 // 512
    const size_t BTD = (size_t)Bsz * T * D;   // 4,194,304

    char* w = (char*)d_ws;
    char* wend = w + ws_size;
    auto carve = [&](size_t bytes) -> void* {
        void* p = (void*)w;
        w += (bytes + 255) & ~(size_t)255;
        return p;
    };
    uint16_t* x_bf      = (uint16_t*)carve(BTD * 2);
    uint16_t* Wqkv_t    = (uint16_t*)carve((size_t)D * 3 * D * 2);   // all weights bf16, TRANSPOSED (N x K)
    uint16_t* Wo_t      = (uint16_t*)carve((size_t)D * D * 2);
    uint16_t* Wc_t      = (uint16_t*)carve((size_t)4 * D * D * 2);
    uint16_t* Wq_t      = (uint16_t*)carve((size_t)D * D * 2);
    uint16_t* Wk_t      = (uint16_t*)carve((size_t)D * D * 2);
    uint16_t* Wv_t      = (uint16_t*)carve((size_t)D * D * 2);
    uint16_t* Wbeta_t   = (uint16_t*)carve((size_t)D * D * 2);
    uint16_t* Walpha_t  = (uint16_t*)carve((size_t)D * D * 2);
    uint16_t* Wg_t      = (uint16_t*)carve((size_t)2 * D * D * 2);
    uint16_t* Wout_t    = (uint16_t*)carve((size_t)D * D * 2);
    uint16_t* qkv_bf    = (uint16_t*)carve(BTD * 3 * 2);
    uint16_t* attnp_bf  = (uint16_t*)carve(BTD * 2);
    float*    attn_f    = (float*)   carve(BTD * 4);
    uint16_t* concat_bf = (uint16_t*)carve(BTD * 2 * 2);
    uint16_t* xc_bf     = (uint16_t*)carve((size_t)Bsz * Tp * D * 2);
    float*    qp        = (float*)   carve((size_t)Bsz * Tp * D * 4);
    float*    kp        = (float*)   carve((size_t)Bsz * Tp * D * 4);
    float*    vp        = (float*)   carve((size_t)Bsz * Tp * D * 4);
    float*    betap     = (float*)   carve((size_t)Bsz * Tp * D * 4);
    float*    logap     = (float*)   carve((size_t)Bsz * Tp * D * 4);
    float*    memsmall  = (float*)   carve((size_t)Bsz * Tp * D * 4);
    float*    mem_f     = (float*)   carve(BTD * 4);
    float*    gatel     = (float*)   carve(BTD * 4);
    uint16_t* comb_bf   = (uint16_t*)carve(BTD * 2);
    if (w > wend) return;   // workspace too small; deterministic no-op

    auto cvt = [&](const float* src, uint16_t* dst, size_t n) {
        k_convert<<<dim3((unsigned)((n + 255) / 256)), dim3(256), 0, stream>>>(src, dst, n);
    };
    auto cvtT = [&](const float* src, uint16_t* dst, int K, int N) {  // src is K x N
        k_convert_t<<<dim3(K / 32, N / 32), dim3(256), 0, stream>>>(src, dst, K, N);
    };
    auto gemm = [&](const uint16_t* A, const uint16_t* Bt, const float* bias,
                    float* outF, uint16_t* outBF,
                    int M, int N, int K, int lda, int ldcF, int ldcBF, int act) {
        k_gemm<<<dim3(N / 64, M / 64), dim3(256), 0, stream>>>(
            A, Bt, bias, outF, outBF, M, N, K, lda, /*ldb=*/K, ldcF, ldcBF, act);
    };

    // 0) bf16 conversions (weights transposed to N x K for the TDM/WMMA fast path)
    cvt (x, x_bf, BTD);
    cvtT(Wqkv,   Wqkv_t,   D,     3 * D);
    cvtT(Wo,     Wo_t,     D,     D);
    cvtT(Wc,     Wc_t,     4 * D, D);
    cvtT(Wq,     Wq_t,     D,     D);
    cvtT(Wk,     Wk_t,     D,     D);
    cvtT(Wv,     Wv_t,     D,     D);
    cvtT(Wbeta,  Wbeta_t,  D,     D);
    cvtT(Walpha, Walpha_t, D,     D);
    cvtT(Wg,     Wg_t,     2 * D, D);
    cvtT(Wout,   Wout_t,   D,     D);

    // 1) QKV projection -> bf16
    gemm(x_bf, Wqkv_t, bqkv, nullptr, qkv_bf, Bsz * T, 3 * D, D, D, 0, 3 * D, 0);

    // 2) sliding-window attention (WMMA flash)
    {
        int waves = Bsz * H * (T / 16);
        k_attn<<<dim3(waves / 4), dim3(128), 0, stream>>>(qkv_bf, attnp_bf, T, H, 3 * D, D);
    }

    // 3) attention out-projection -> f32 + bf16 into first half of concat
    gemm(attnp_bf, Wo_t, bo, attn_f, concat_bf, Bsz * T, D, D, D, D, 2 * D, 0);

    // 4) xc = reshape(x, (B, T/4, 4D)) @ Wc + bc -> bf16
    gemm(x_bf, Wc_t, bcv, nullptr, xc_bf, Bsz * Tp, D, 4 * D, 4 * D, 0, D, 0);

    // 5) delta projections (fused activations)
    gemm(xc_bf, Wq_t,     bq,     qp,    nullptr, Bsz * Tp, D, D, D, D, 0, 1); // *0.125
    gemm(xc_bf, Wk_t,     bk,     kp,    nullptr, Bsz * Tp, D, D, D, D, 0, 1); // *0.125
    gemm(xc_bf, Wv_t,     bv,     vp,    nullptr, Bsz * Tp, D, D, D, D, 0, 0);
    gemm(xc_bf, Wbeta_t,  bbeta,  betap, nullptr, Bsz * Tp, D, D, D, D, 0, 2); // softplus
    gemm(xc_bf, Walpha_t, balpha, logap, nullptr, Bsz * Tp, D, D, D, D, 0, 3); // log_sigmoid(clip)

    // 6) chunked delta scan; writes memsmall and next_state (tail of d_out)
    k_delta<<<dim3(Bsz * H), dim3(256), 0, stream>>>(
        qp, kp, vp, betap, logap, state, memsmall, (float*)d_out + BTD, Tp, H, D);

    // 7) mem = roll(repeat(memsmall, 4), 4); first 4 steps zero; bf16 half of concat
    k_expand_mem<<<dim3((unsigned)((BTD + 255) / 256)), dim3(256), 0, stream>>>(
        memsmall, mem_f, concat_bf, Bsz, T, D, R);

    // 8) gate logits
    gemm(concat_bf, Wg_t, bg, gatel, nullptr, Bsz * T, D, 2 * D, 2 * D, D, 0, 0);

    // 9) combined = sigmoid(gate)*attn + (1-sigmoid)*mem -> bf16
    k_combine<<<dim3((unsigned)((BTD + 255) / 256)), dim3(256), 0, stream>>>(
        gatel, attn_f, mem_f, comb_bf, BTD);

    // 10) out = combined @ Wout + bout -> d_out (f32)
    gemm(comb_bf, Wout_t, bout, (float*)d_out, nullptr, Bsz * T, D, D, D, D, 0, 0);
}
